// GCNModel_52012053955147
// MI455X (gfx1250) — matrix operand order repaired and checked by reading, not compiled, not run on hardware
//
#include <hip/hip_runtime.h>
#include <hip/hip_bf16.h>
#include <math.h>

// ---------------------------------------------------------------------------
// GCN on MI455X (gfx1250, wave32).
//   h1 = relu(GCNConv(x, W1, b1)); h2 = GCNConv(h1, W2, b2)
//   out[b,n,d] = log_softmax over n of (h2[n,d] + q[b,d])
//              = h2[n,d] - max_n h2[:,d] - log sum_n exp(...)   (q cancels)
// GEMMs: v_wmma_f32_16x16x32_bf16, A tile staged via TENSOR_LOAD_TO_LDS (TDM)
// with DMA-side LDS padding for conflict-free ds_load_b128 fragment reads.
// ---------------------------------------------------------------------------

#define D 128
#define ASTRIDE 132   // A row stride in LDS (floats): TDM pads 4 dwords / 128
#define WSTRIDE 136   // W^T row stride in LDS (bf16): 272B -> bank stride 4

typedef __bf16 v16bf __attribute__((ext_vector_type(16)));
typedef __bf16 v8bf  __attribute__((ext_vector_type(8)));
typedef float  v8f   __attribute__((ext_vector_type(8)));
typedef unsigned int u32x4 __attribute__((ext_vector_type(4)));
typedef int          i32x8 __attribute__((ext_vector_type(8)));
typedef int          i32x4 __attribute__((ext_vector_type(4)));

#if __has_builtin(__builtin_amdgcn_tensor_load_to_lds) && \
    __has_builtin(__builtin_amdgcn_s_wait_tensorcnt)
#define HAVE_TDM 1
#else
#define HAVE_TDM 0
#endif

// Native bf16 convert: lowers to v_cvt_*bf16_f32 (RNE) instead of a
// 3-op software rounding chain per element.
__device__ __forceinline__ __bf16 f2bf(float f) { return (__bf16)f; }

// ---------------------------------------------------------------------------
// Degree / rsqrt
// ---------------------------------------------------------------------------
__global__ void degree_kernel(const int* __restrict__ cols, float* __restrict__ deg, int n_edges) {
    int e = blockIdx.x * blockDim.x + threadIdx.x;
    if (e < n_edges) {
        __builtin_prefetch(&cols[e + 8192], 0, 1);   // global_prefetch_b8
        atomicAdd(&deg[cols[e]], 1.0f);
    }
}

__global__ void dinv_kernel(float* __restrict__ deg, int n) {
    int i = blockIdx.x * blockDim.x + threadIdx.x;
    if (i < n) deg[i] = rsqrtf(deg[i] + 1.0f);       // +1 self loop
}

// ---------------------------------------------------------------------------
// out[n,0:128] = A[n,0:128] @ W[128,128]  via WMMA bf16.
// Block = 256 threads = 8 waves, covers 128 rows. A tile (128x128 f32) is
// DMA'd into LDS by the Tensor Data Mover with a 4-dword row pad; W is staged
// transposed as bf16 so every fragment read is a pair of ds_load_b128.
// `out` has >= rowStart+128 rows of backing store (padded workspace), so the
// C store loop is unconditional; TDM's tensor_dim1 clamp zero-fills A rows
// past n_rows.
// Dynamic LDS: [ A: 128*132*4 = 67584 B ][ W^T: 128*136*2 = 34816 B ]
// ---------------------------------------------------------------------------
__global__ __launch_bounds__(256) void gemm_wmma_kernel(const float* __restrict__ A,
                                                        const float* __restrict__ W,
                                                        float* __restrict__ out,
                                                        int n_rows) {
    extern __shared__ __align__(16) char smem[];
    float*  Al = (float*)smem;                              // [128][ASTRIDE]
    __bf16* Wt = (__bf16*)(smem + 128 * ASTRIDE * 4);       // [128][WSTRIDE]

    const int tid = threadIdx.x;
    const int rowStart = blockIdx.x * 128;
    const int rowsHere = min(128, n_rows - rowStart);

    // ---- stage W^T -> LDS as bf16 (coalesced read, scattered LDS write) ----
    for (int i = tid; i < D * D; i += 256) {
        int k = i >> 7, n = i & 127;
        Wt[n * WSTRIDE + k] = f2bf(W[i]);
    }

    // ---- stage A tile -> LDS via TDM (async tensor DMA, padded rows) ----
#if HAVE_TDM
    if (tid < 32) {
        unsigned lds_addr = (unsigned)(unsigned long long)(void*)Al;
        unsigned long long ga = (unsigned long long)(const void*)(A + (size_t)rowStart * D);
        u32x4 g0;
        g0[0] = 1u;                                   // count=1, user mode
        g0[1] = lds_addr;                             // LDS destination
        g0[2] = (unsigned)ga;                         // global_addr[31:0]
        g0[3] = ((unsigned)(ga >> 32) & 0x01FFFFFFu)  // global_addr[56:32]
                | (2u << 30);                         // type = 2 (image)
        i32x8 g1;
        g1[0] = (int)((2u << 16)      // data_size = 4 bytes
                    | (1u << 20)      // pad_enable
                    | (6u << 22)      // pad_interval: every 128 dwords
                    | (3u << 25));    // pad_amount: 4 dwords -> row stride 132
        g1[1] = (int)(128u << 16);    // tensor_dim0 = 128 elements
        g1[2] = (int)(((unsigned)rowsHere & 0xFFFFu) << 16); // tensor_dim1 (OOB rows -> 0)
        g1[3] = (int)(128u << 16);    // tile_dim0 = 128
        g1[4] = 128;                  // tile_dim1 = 128
        g1[5] = 128;                  // tensor_dim0_stride = 128 elements
        g1[6] = 0;
        g1[7] = 0;
        i32x4 z4 = (i32x4){0, 0, 0, 0};
#if defined(__clang_major__) && __clang_major__ >= 23
        i32x8 z8 = (i32x8){0, 0, 0, 0, 0, 0, 0, 0};
        __builtin_amdgcn_tensor_load_to_lds(g0, g1, z4, z4, z8, 0);
#else
        __builtin_amdgcn_tensor_load_to_lds(g0, g1, z4, z4, 0);
#endif
        __builtin_amdgcn_s_wait_tensorcnt(0);
    }
#else
    for (int i = tid; i < 128 * D; i += 256) {
        int r = i >> 7, k = i & 127;
        float v = (r < rowsHere) ? A[(size_t)(rowStart + r) * D + k] : 0.0f;
        Al[r * ASTRIDE + k] = v;
    }
#endif
    __syncthreads();

    const int wave = tid >> 5;
    const int lane = tid & 31;
    const int m    = lane & 15;          // row-in-tile / col-in-tile
    const int hi   = lane >> 4;          // lane half select
    const float* arow = Al + (size_t)(wave * 16 + m) * ASTRIDE;

    v8f acc[8] = {};                     // 8 column tiles of 16x16 f32

    #pragma unroll
    for (int kk = 0; kk < D; kk += 32) {
        // A fragment (16-bit A 16x32 layout): elem i -> K = kk+(i/8)*16+hi*8+(i%8)
        const float4 a0 = *(const float4*)(arow + kk + (hi << 3));
        const float4 a1 = *(const float4*)(arow + kk + (hi << 3) + 4);
        const float4 a2 = *(const float4*)(arow + kk + 16 + (hi << 3));
        const float4 a3 = *(const float4*)(arow + kk + 16 + (hi << 3) + 4);
        v16bf afrag;
        afrag[0]  = f2bf(a0.x); afrag[1]  = f2bf(a0.y); afrag[2]  = f2bf(a0.z); afrag[3]  = f2bf(a0.w);
        afrag[4]  = f2bf(a1.x); afrag[5]  = f2bf(a1.y); afrag[6]  = f2bf(a1.z); afrag[7]  = f2bf(a1.w);
        afrag[8]  = f2bf(a2.x); afrag[9]  = f2bf(a2.y); afrag[10] = f2bf(a2.z); afrag[11] = f2bf(a2.w);
        afrag[12] = f2bf(a3.x); afrag[13] = f2bf(a3.y); afrag[14] = f2bf(a3.z); afrag[15] = f2bf(a3.w);

        const int kb = kk + (hi << 3);   // B: elem i -> K = kk+(i/8)*16+hi*8+(i%8)
        #pragma unroll
        for (int t = 0; t < 8; ++t) {
            const __bf16* bp = Wt + ((t << 4) + m) * WSTRIDE + kb;   // two runs of 8
            union { v16bf v; v8bf h[2]; } bu;
            bu.h[0] = *(const v8bf*)bp;          // ds_load_b128
            bu.h[1] = *(const v8bf*)(bp + 16);   // ds_load_b128
            acc[t] = __builtin_amdgcn_wmma_f32_16x16x32_bf16(
                false, afrag, false, bu.v, (short)0, acc[t], false, false);
        }
    }

    // C/D layout: VGPR v -> M = v + hi*8, N = lane%16.  Unconditional store
    // into the 128-row-padded output buffer (no per-element bounds checks).
    const int mbase = rowStart + wave * 16 + (hi << 3);
    #pragma unroll
    for (int t = 0; t < 8; ++t) {
        #pragma unroll
        for (int v = 0; v < 8; ++v) {
            out[(size_t)(mbase + v) * D + (t << 4) + m] = acc[t][v];
        }
    }
}

// ---------------------------------------------------------------------------
// Edge aggregation: one wave per edge, each lane moves 4 floats (b128 load,
// 4x global_atomic_add_f32).  agg[col] += xw[row] * dinv[row]*dinv[col]
// ---------------------------------------------------------------------------
__global__ __launch_bounds__(256) void aggregate_kernel(const int* __restrict__ rows,
                                                        const int* __restrict__ cols,
                                                        const float* __restrict__ dinv,
                                                        const float* __restrict__ xw,
                                                        float* __restrict__ agg,
                                                        int n_edges) {
    const int e    = (int)(((size_t)blockIdx.x * blockDim.x + threadIdx.x) >> 5);
    const int lane = threadIdx.x & 31;
    if (e >= n_edges) return;
    __builtin_prefetch(&rows[e + 2048], 0, 1);
    __builtin_prefetch(&cols[e + 2048], 0, 1);
    const int r = rows[e], c = cols[e];
    const float norm = dinv[r] * dinv[c];
    const float4 v = ((const float4*)(xw + (size_t)r * D))[lane];
    float* dst = agg + (size_t)c * D + lane * 4;
    atomicAdd(dst + 0, v.x * norm);
    atomicAdd(dst + 1, v.y * norm);
    atomicAdd(dst + 2, v.z * norm);
    atomicAdd(dst + 3, v.w * norm);
}

// ---------------------------------------------------------------------------
// h = (agg + dinv^2 * xw + b)  [optional relu]
// ---------------------------------------------------------------------------
__global__ void finalize_kernel(const float* __restrict__ agg, const float* __restrict__ xw,
                                const float* __restrict__ dinv, const float* __restrict__ bias,
                                float* __restrict__ out, int n, int do_relu) {
    size_t idx = (size_t)blockIdx.x * blockDim.x + threadIdx.x;
    if (idx >= (size_t)n * D) return;
    int node = (int)(idx >> 7), d = (int)(idx & 127);
    float di = dinv[node];
    float v  = agg[idx] + di * di * xw[idx] + bias[d];
    if (do_relu) v = fmaxf(v, 0.0f);
    out[idx] = v;
}

// ---------------------------------------------------------------------------
// q = qemb @ Wq + bq   (4x128 — trivial)
// ---------------------------------------------------------------------------
__global__ void qproj_kernel(const float* __restrict__ qe, const float* __restrict__ Wq,
                             const float* __restrict__ bq, float* __restrict__ q) {
    int t = threadIdx.x;           // 0..511
    int b = t >> 7, d = t & 127;
    float s = bq[d];
    #pragma unroll 8
    for (int k = 0; k < D; ++k) s += qe[b * D + k] * Wq[k * D + d];
    q[t] = s;
}

// ---------------------------------------------------------------------------
// Per-column max and log-sum-exp over the node dimension (softmax axis).
// ---------------------------------------------------------------------------
__global__ __launch_bounds__(256) void colstats_kernel(const float* __restrict__ h, int n,
                                                       float* __restrict__ Md,
                                                       float* __restrict__ lSd) {
    __shared__ float red[256];
    const int d = blockIdx.x, t = threadIdx.x;
    float m = -INFINITY;
    for (int i = t; i < n; i += 256) m = fmaxf(m, h[(size_t)i * D + d]);
    red[t] = m; __syncthreads();
    for (int s = 128; s > 0; s >>= 1) { if (t < s) red[t] = fmaxf(red[t], red[t + s]); __syncthreads(); }
    const float mx = red[0]; __syncthreads();
    float sum = 0.0f;
    for (int i = t; i < n; i += 256) sum += __expf(h[(size_t)i * D + d] - mx);
    red[t] = sum; __syncthreads();
    for (int s = 128; s > 0; s >>= 1) { if (t < s) red[t] += red[t + s]; __syncthreads(); }
    if (t == 0) { Md[d] = mx; lSd[d] = logf(red[0]); }
}

// ---------------------------------------------------------------------------
// out[b,n,d] = h[n,d] - Md[d] - lSd[d]   (q cancels; broadcast over b)
// ---------------------------------------------------------------------------
__global__ void writeout_kernel(const float* __restrict__ h, const float* __restrict__ Md,
                                const float* __restrict__ lSd, float* __restrict__ out, int n) {
    size_t idx = (size_t)blockIdx.x * blockDim.x + threadIdx.x;
    if (idx >= (size_t)n * D) return;
    int d = (int)(idx & 127);
    float v = h[idx] - Md[d] - lSd[d];
    size_t stride = (size_t)n * D;
    #pragma unroll
    for (int b = 0; b < 4; ++b) out[b * stride + idx] = v;
}

// ---------------------------------------------------------------------------
extern "C" void kernel_launch(void* const* d_in, const int* in_sizes, int n_in,
                              void* d_out, int out_size, void* d_ws, size_t ws_size,
                              hipStream_t stream) {
    const float* x    = (const float*)d_in[0];
    const int*   ei   = (const int*)  d_in[1];
    const float* qemb = (const float*)d_in[2];
    const float* W1   = (const float*)d_in[3];
    const float* b1   = (const float*)d_in[4];
    const float* W2   = (const float*)d_in[5];
    const float* b2   = (const float*)d_in[6];
    const float* Wq   = (const float*)d_in[7];
    const float* bq   = (const float*)d_in[8];

    const int n_nodes = in_sizes[0] / D;
    const int n_edges = in_sizes[1] / 2;
    const int* rows = ei;
    const int* cols = ei + n_edges;

    // Workspace: pad [N,D] buffers to a 128-row multiple so the WMMA GEMM can
    // store its full 128-row tile unconditionally.
    const int n_pad = (n_nodes + 127) & ~127;
    float* ws   = (float*)d_ws;
    size_t nd   = (size_t)n_nodes * D;
    size_t ndp  = (size_t)n_pad  * D;
    float* xw   = ws;                 // [n_pad,D]
    float* agg  = ws + ndp;           // [n_pad,D]
    float* h    = ws + 2 * ndp;       // [n_pad,D]
    float* dinv = ws + 3 * ndp;       // [N]
    float* q    = dinv + n_nodes;     // [4,128]
    float* Md   = q + 4 * D;          // [128]
    float* lSd  = Md + D;             // [128]

    const int T = 256;
    const int blkE   = (n_edges + T - 1) / T;
    const int blkEW  = (int)(((size_t)n_edges * 32 + T - 1) / T);
    const int blkN   = (n_nodes + T - 1) / T;
    const int blkND  = (int)((nd + T - 1) / T);
    const int blkGem = n_pad / 128;
    const size_t gemmLds = (size_t)128 * ASTRIDE * 4 + (size_t)128 * WSTRIDE * 2; // 102400 B

    // ---- degrees -> dinv ----
    hipMemsetAsync(dinv, 0, (size_t)n_nodes * sizeof(float), stream);
    degree_kernel<<<blkE, T, 0, stream>>>(cols, dinv, n_edges);
    dinv_kernel<<<blkN, T, 0, stream>>>(dinv, n_nodes);

    // ---- layer 1 ----
    gemm_wmma_kernel<<<blkGem, T, gemmLds, stream>>>(x, W1, xw, n_nodes);
    hipMemsetAsync(agg, 0, nd * sizeof(float), stream);
    aggregate_kernel<<<blkEW, T, 0, stream>>>(rows, cols, dinv, xw, agg, n_edges);
    finalize_kernel<<<blkND, T, 0, stream>>>(agg, xw, dinv, b1, h, n_nodes, 1);

    // ---- layer 2 ----
    gemm_wmma_kernel<<<blkGem, T, gemmLds, stream>>>(h, W2, xw, n_nodes);
    hipMemsetAsync(agg, 0, nd * sizeof(float), stream);
    aggregate_kernel<<<blkEW, T, 0, stream>>>(rows, cols, dinv, xw, agg, n_edges);
    finalize_kernel<<<blkND, T, 0, stream>>>(agg, xw, dinv, b2, h, n_nodes, 0);

    // ---- question projection (cancels in log_softmax; kept for fidelity) ----
    qproj_kernel<<<1, 512, 0, stream>>>(qemb, Wq, bq, q);

    // ---- log_softmax over node dim ----
    colstats_kernel<<<D, T, 0, stream>>>(h, n_nodes, Md, lSd);
    writeout_kernel<<<blkND, T, 0, stream>>>(h, Md, lSd, (float*)d_out, n_nodes);
}
